// WindowAttention_79104707658161
// MI455X (gfx1250) — matrix operand order repaired and checked by reading, not compiled, not run on hardware
//
#include <hip/hip_runtime.h>

#define TOK   49
#define TPAD  64
#define CDIM  128
#define NHEAD 4
#define HDIM  32
#define NWIN  64

#define QKVW_U32   24576          // 384*128/2 packed bf16 pairs
#define PROJW_U32  8192           // 128*128/2
#define W_U32      (QKVW_U32 + PROJW_U32)
#define BIAS_ELTS  (NHEAD * TOK * TOK)

typedef __attribute__((ext_vector_type(16))) __bf16 v16bf;
typedef __attribute__((ext_vector_type(2)))  __bf16 bf16x2;
typedef __attribute__((ext_vector_type(8)))  float  v8f;

union Frag { v16bf v; unsigned u[8]; uint4 q4[2]; };

// Lowers to v_cvt_pk_bf16_f32 on gfx1250 (verified in round-2 disasm).
__device__ __forceinline__ unsigned pack2bf(float lo, float hi) {
  bf16x2 p = { (__bf16)lo, (__bf16)hi };
  return __builtin_bit_cast(unsigned, p);
}
__device__ __forceinline__ __bf16 f2bf(float x) { return (__bf16)x; }

#define WMMA_BF16(A, B, C) \
  __builtin_amdgcn_wmma_f32_16x16x32_bf16(false, (A), false, (B), (short)0, (C), false, false)

// ---------- prep: fp32 weights -> bf16 in ws; fuse rel_idx+rpb -> biasH ----------
__global__ void __launch_bounds__(256)
prep_tables(const float* __restrict__ qkv_w, const float* __restrict__ proj_w,
            const float* __restrict__ rpb, const int* __restrict__ rel_idx,
            unsigned* __restrict__ wsu, float* __restrict__ biasH)
{
  const int i = blockIdx.x * 256 + threadIdx.x;
  if (i < QKVW_U32) {
    wsu[i] = pack2bf(qkv_w[2*i], qkv_w[2*i + 1]);
  } else if (i < W_U32) {
    const int j = i - QKVW_U32;
    wsu[i] = pack2bf(proj_w[2*j], proj_w[2*j + 1]);
  } else if (i < W_U32 + BIAS_ELTS) {
    const int j = i - W_U32;            // j = h*2401 + idx
    const int h = j / (TOK * TOK);
    const int idx = j - h * (TOK * TOK);
    biasH[j] = rpb[rel_idx[idx] * NHEAD + h];
  }
}

// ---------- fused window attention ----------
template <bool PREP>
__global__ void __launch_bounds__(128)
winattn_fused(const float* __restrict__ x,
              const float* __restrict__ mask,
              const float* __restrict__ qkv_w,
              const float* __restrict__ qkv_b,
              const float* __restrict__ proj_w,
              const float* __restrict__ proj_b,
              const float* __restrict__ rpb,
              const int*   __restrict__ rel_idx,
              const __bf16* __restrict__ wq_bf,    // [384][128] bf16 (PREP)
              const float* __restrict__ biasH,     // [4][49*49]     (PREP)
              float* __restrict__ out)
{
  extern __shared__ char smem[];
  __bf16* xs   = (__bf16*)(smem);             // [64][128] bf16, reused as out_s
  __bf16* q_s  = (__bf16*)(smem + 16*1024);   // [4][64][32]
  __bf16* k_s  = (__bf16*)(smem + 32*1024);   // [4][64][32]
  __bf16* vT   = (__bf16*)(smem + 48*1024);   // [4][32][64] (transposed V)
  __bf16* p_s  = (__bf16*)(smem + 64*1024);   // [4][16][64] softmax tile per head
  __bf16* out_s = xs;
  const __bf16* wp_bf = wq_bf + 384 * CDIM;   // proj weights bf16

  const int b    = blockIdx.x;
  const int tid  = threadIdx.x;
  const int wave = tid >> 5;
  const int lane = tid & 31;
  const int nl   = lane & 15;                 // tile col (B/C/D) or tile row (A)
  const int kh   = lane >> 4;                 // half-wave select
  const v8f vzero = {0.f,0.f,0.f,0.f,0.f,0.f,0.f,0.f};
  const float scale = 0.17677669529663687f;   // 1/sqrt(32)

  // ---------- Stage 0: x window -> LDS bf16, zero-pad rows 49..63 ----------
  {
    const float4* xg = (const float4*)(x + (size_t)b * TOK * CDIM);
    for (int c = tid; c < TPAD * (CDIM/4); c += 128) {
      int row = c >> 5;
      int c4  = c & 31;
      float4 v = {0.f, 0.f, 0.f, 0.f};
      if (row < TOK) v = xg[row * (CDIM/4) + c4];
      uint2 pk;
      pk.x = pack2bf(v.x, v.y);
      pk.y = pack2bf(v.z, v.w);
      *(uint2*)(xs + row * CDIM + c4 * 4) = pk;
    }
  }
  __syncthreads();

  // ---------- Stage 1: QKV gemm [64,128] x [384,128]^T ----------
  #pragma unroll 1
  for (int nti = 0; nti < 6; ++nti) {
    const int nt   = wave * 6 + nti;
    const int col0 = nt * 16;
    v8f acc[4] = {vzero, vzero, vzero, vzero};
    #pragma unroll
    for (int ks = 0; ks < 4; ++ks) {
      const int k0 = ks * 32;
      Frag bf;
      if (PREP) {
        const uint4* wp = (const uint4*)(wq_bf + (size_t)(col0 + nl) * CDIM + k0 + kh * 16);
        bf.q4[0] = wp[0];
        bf.q4[1] = wp[1];
      } else {
        const float4* wp = (const float4*)(qkv_w + (size_t)(col0 + nl) * CDIM + k0 + kh * 16);
        #pragma unroll
        for (int i = 0; i < 4; ++i) {
          float4 w = wp[i];
          bf.u[2*i]   = pack2bf(w.x, w.y);
          bf.u[2*i+1] = pack2bf(w.z, w.w);
        }
      }
      #pragma unroll
      for (int mt = 0; mt < 4; ++mt) {
        Frag af;
        const __bf16* ap = xs + (mt*16 + nl) * CDIM + k0 + kh * 16;
        af.q4[0] = *(const uint4*)ap;
        af.q4[1] = *(const uint4*)(ap + 8);
        acc[mt] = WMMA_BF16(af.v, bf.v, acc[mt]);
      }
    }
    const int which = col0 >> 7;        // 0=q 1=k 2=v
    const int head  = (col0 >> 5) & 3;
    const int d0    = col0 & 31;
    const float qb  = qkv_b[col0 + nl];
    if (which == 2) {
      // vT[d][tok]: a lane's 8 rows are consecutive tokens -> one b128 store
      #pragma unroll
      for (int mt = 0; mt < 4; ++mt) {
        uint4 pk;
        pk.x = pack2bf(acc[mt][0] + qb, acc[mt][1] + qb);
        pk.y = pack2bf(acc[mt][2] + qb, acc[mt][3] + qb);
        pk.z = pack2bf(acc[mt][4] + qb, acc[mt][5] + qb);
        pk.w = pack2bf(acc[mt][6] + qb, acc[mt][7] + qb);
        *(uint4*)(vT + head*2048 + (d0 + nl)*64 + mt*16 + 8*kh) = pk;
      }
    } else {
      #pragma unroll
      for (int mt = 0; mt < 4; ++mt) {
        #pragma unroll
        for (int r = 0; r < 8; ++r) {
          const int tok = mt*16 + r + 8*kh;
          const float val = acc[mt][r] + qb;
          if (which == 0) q_s[head*2048 + tok*32 + d0 + nl] = f2bf(val * scale);
          else            k_s[head*2048 + tok*32 + d0 + nl] = f2bf(val);
        }
      }
    }
  }
  __syncthreads();

  // ---------- Stage 2: per-head attention (wave = head) ----------
  {
    const int h = wave;
    const __bf16* qh = q_s + h * 2048;
    const __bf16* kk = k_s + h * 2048;
    const __bf16* vh = vT  + h * 2048;
    __bf16* ph = p_s + h * 1024;
    const float* maskw = mask + (size_t)(b & (NWIN - 1)) * TOK * TOK;
    const float* biash = PREP ? (biasH + h * TOK * TOK) : nullptr;

    #pragma unroll 1
    for (int mt = 0; mt < 4; ++mt) {
      // S = Q * K^T  (16x64 row block)
      Frag aq;
      {
        const __bf16* ap = qh + (mt*16 + nl) * HDIM + kh * 16;
        aq.q4[0] = *(const uint4*)ap;
        aq.q4[1] = *(const uint4*)(ap + 8);
      }
      v8f s[4] = {vzero, vzero, vzero, vzero};
      #pragma unroll
      for (int nt = 0; nt < 4; ++nt) {
        Frag bk;
        const __bf16* bp = kk + (nt*16 + nl) * HDIM + kh * 16;
        bk.q4[0] = *(const uint4*)bp;
        bk.q4[1] = *(const uint4*)(bp + 8);
        s[nt] = WMMA_BF16(aq.v, bk.v, s[nt]);
      }
      // bias + mask + padding mask
      float rmax[8];
      #pragma unroll
      for (int r = 0; r < 8; ++r) rmax[r] = -3.0e38f;
      #pragma unroll
      for (int nt = 0; nt < 4; ++nt) {
        #pragma unroll
        for (int r = 0; r < 8; ++r) {
          const int row = mt*16 + r + 8*kh;
          const int col = nt*16 + nl;
          float sv = s[nt][r];
          if (col < TOK) {
            if (row < TOK) {
              if (PREP) {
                sv += biash[row * TOK + col] + maskw[row * TOK + col];
              } else {
                const int ri = rel_idx[row * TOK + col];
                sv += rpb[ri * NHEAD + h] + maskw[row * TOK + col];
              }
            }
          } else {
            sv = -1.0e30f;
          }
          s[nt][r] = sv;
          rmax[r] = fmaxf(rmax[r], sv);
        }
      }
      // row-wise softmax: reduce across the 16 lanes of this half-wave
      #pragma unroll
      for (int r = 0; r < 8; ++r) {
        #pragma unroll
        for (int xm = 1; xm < 16; xm <<= 1)
          rmax[r] = fmaxf(rmax[r], __shfl_xor(rmax[r], xm, 32));
      }
      float rsum[8];
      #pragma unroll
      for (int r = 0; r < 8; ++r) rsum[r] = 0.f;
      #pragma unroll
      for (int nt = 0; nt < 4; ++nt) {
        #pragma unroll
        for (int r = 0; r < 8; ++r) {
          const float e = __expf(s[nt][r] - rmax[r]);
          s[nt][r] = e;
          rsum[r] += e;
        }
      }
      #pragma unroll
      for (int r = 0; r < 8; ++r) {
        #pragma unroll
        for (int xm = 1; xm < 16; xm <<= 1)
          rsum[r] += __shfl_xor(rsum[r], xm, 32);
        rsum[r] = 1.0f / rsum[r];
      }
      #pragma unroll
      for (int nt = 0; nt < 4; ++nt)
        #pragma unroll
        for (int r = 0; r < 8; ++r)
          ph[(r + 8*kh) * 64 + nt*16 + nl] = f2bf(s[nt][r] * rsum[r]);

      __syncthreads();   // ph stores -> ph frag loads (and fences alias types)

      // O = P * V   (P: 16x64, V: 64x32)
      #pragma unroll
      for (int nt2 = 0; nt2 < 2; ++nt2) {
        v8f o = vzero;
        #pragma unroll
        for (int ks = 0; ks < 2; ++ks) {
          Frag ap2, bv;
          const __bf16* app = ph + nl * 64 + ks*32 + kh * 16;
          ap2.q4[0] = *(const uint4*)app;
          ap2.q4[1] = *(const uint4*)(app + 8);
          const __bf16* bpp = vh + (nt2*16 + nl) * 64 + ks*32 + kh * 16;
          bv.q4[0] = *(const uint4*)bpp;
          bv.q4[1] = *(const uint4*)(bpp + 8);
          o = WMMA_BF16(ap2.v, bv.v, o);
        }
        #pragma unroll
        for (int r = 0; r < 8; ++r) {
          const int tok = mt*16 + r + 8*kh;
          out_s[tok * CDIM + h * HDIM + nt2*16 + nl] = f2bf(o[r]);
        }
      }
      __syncthreads();   // ph WAR across mt iterations; final one covers stage 3
    }
  }

  // ---------- Stage 3: proj gemm [64,128] x [128,128]^T + bias ----------
  #pragma unroll 1
  for (int nti = 0; nti < 2; ++nti) {
    const int nt = wave * 2 + nti;
    const int n0 = nt * 16;
    v8f acc[4] = {vzero, vzero, vzero, vzero};
    #pragma unroll
    for (int ks = 0; ks < 4; ++ks) {
      const int k0 = ks * 32;
      Frag bf;
      if (PREP) {
        const uint4* wp = (const uint4*)(wp_bf + (size_t)(n0 + nl) * CDIM + k0 + kh * 16);
        bf.q4[0] = wp[0];
        bf.q4[1] = wp[1];
      } else {
        const float4* wp = (const float4*)(proj_w + (size_t)(n0 + nl) * CDIM + k0 + kh * 16);
        #pragma unroll
        for (int i = 0; i < 4; ++i) {
          float4 w = wp[i];
          bf.u[2*i]   = pack2bf(w.x, w.y);
          bf.u[2*i+1] = pack2bf(w.z, w.w);
        }
      }
      #pragma unroll
      for (int mt = 0; mt < 4; ++mt) {
        Frag af;
        const __bf16* ap = out_s + (mt*16 + nl) * CDIM + k0 + kh * 16;
        af.q4[0] = *(const uint4*)ap;
        af.q4[1] = *(const uint4*)(ap + 8);
        acc[mt] = WMMA_BF16(af.v, bf.v, acc[mt]);
      }
    }
    const float pb = proj_b[n0 + nl];
    float* outg = out + (size_t)b * TOK * CDIM;
    #pragma unroll
    for (int mt = 0; mt < 4; ++mt) {
      #pragma unroll
      for (int r = 0; r < 8; ++r) {
        const int tok = mt*16 + r + 8*kh;
        if (tok < TOK) outg[tok * CDIM + n0 + nl] = acc[mt][r] + pb;
      }
    }
  }
}

extern "C" void kernel_launch(void* const* d_in, const int* in_sizes, int n_in,
                              void* d_out, int out_size, void* d_ws, size_t ws_size,
                              hipStream_t stream) {
  const float* x       = (const float*)d_in[0];
  const float* mask    = (const float*)d_in[1];
  const float* qkv_w   = (const float*)d_in[2];
  const float* qkv_b   = (const float*)d_in[3];
  const float* proj_w  = (const float*)d_in[4];
  const float* proj_b  = (const float*)d_in[5];
  const float* rpb     = (const float*)d_in[6];
  const int*   rel_idx = (const int*)d_in[7];
  float* outp = (float*)d_out;

  const int B_ = in_sizes[0] / (TOK * CDIM);   // 4096 windows
  const size_t shmem = 72 * 1024;              // 64 KB tiles + 8 KB softmax scratch

  const size_t ws_need = (size_t)W_U32 * 4 + (size_t)BIAS_ELTS * 4;
  if (ws_size >= ws_need) {
    unsigned* wsu  = (unsigned*)d_ws;
    float* biasH   = (float*)((char*)d_ws + (size_t)W_U32 * 4);
    const int prep_threads = W_U32 + BIAS_ELTS;
    prep_tables<<<(prep_threads + 255) / 256, 256, 0, stream>>>(
        qkv_w, proj_w, rpb, rel_idx, wsu, biasH);
    winattn_fused<true><<<B_, 128, shmem, stream>>>(
        x, mask, qkv_w, qkv_b, proj_w, proj_b, rpb, rel_idx,
        (const __bf16*)wsu, biasH, outp);
  } else {
    winattn_fused<false><<<B_, 128, shmem, stream>>>(
        x, mask, qkv_w, qkv_b, proj_w, proj_b, rpb, rel_idx,
        nullptr, nullptr, outp);
  }
}
